// PMHCI_Network01_30262339568087
// MI455X (gfx1250) — matrix operand order repaired and checked by reading, not compiled, not run on hardware
//
#include <hip/hip_runtime.h>
#include <stdint.h>

// ---------------- problem constants (match reference) ----------------
#define N_NODES 50000
#define N_EDGES 800000
#define D_NODE  64
#define D_EDGE  16
#define MSG     32
#define HID     128
#define N_GRAPHS 500

#define E_IN   144           // 2*D_NODE + D_EDGE
#define E_INP  160           // padded to multiple of 32 (K for WMMA)
#define E_KS   5             // K-steps for edge MLP
#define E_NT   2             // N-tiles (MSG = 32)
#define N_IN   96            // D_NODE + MSG
#define N_KS   3
#define N_NT   4             // N-tiles (D_NODE = 64)

typedef __attribute__((ext_vector_type(16))) _Float16 v16h;
typedef __attribute__((ext_vector_type(4)))  _Float16 v4h;
typedef __attribute__((ext_vector_type(8)))  float    v8f;

// Inverse of the CDNA5 16-bit A-matrix (16x32) fragment layout:
// element (row r, K k) lives at lane/half:
//   kr=k%32: kr<8 -> lane=r,    h=kr
//            kr<16-> lane=16+r, h=kr-8
//            kr<24-> lane=r,    h=8+(kr-16)
//            else -> lane=16+r, h=8+(kr-24)
// Storage: [(ks*32 + lane)*16 + h] halfs  => each lane's fragment is one
// contiguous 32B chunk (2x ds_load_b128).
__device__ __forceinline__ int frag_idx(int r, int k) {
    int ks = k >> 5;
    int kr = k & 31;
    int lane = ((kr & 8) ? 16 : 0) + r;      // kr in [8,16) or [24,32) -> hi lane
    int h    = ((kr & 16) ? 8 : 0) + (kr & 7);
    return (ks * 32 + lane) * 16 + h;
}

// ---------------------------------------------------------------------
// Pack fp32 weights [Kreal x Ncols] into f16 swizzled B-fragment order:
// dst[((ks*NT + nt)*32 + lane)*16 + h],  k = ks*32 + (h>=8?16:0) + (lane>=16?8:0) + h%8,
// n = nt*16 + lane%16.  (B 32x16 layout: lane%16 = col, K=(h>=8?16:0)+(lane<16?0:8)+h%8)
// ---------------------------------------------------------------------
__global__ void pack_w(const float* __restrict__ W, _Float16* __restrict__ dst,
                       int Kreal, int Ncols, int NT, int total)
{
    int i = blockIdx.x * blockDim.x + threadIdx.x;
    if (i >= total) return;
    int h    = i & 15;
    int lane = (i >> 4) & 31;
    int nt   = (i >> 9) % NT;
    int ks   = i / (512 * NT);
    int k = ks * 32 + ((h & 8) ? 16 : 0) + ((lane >= 16) ? 8 : 0) + (h & 7);
    int n = nt * 16 + (lane & 15);
    dst[i] = (k < Kreal) ? (_Float16)W[k * Ncols + n] : (_Float16)0.f;
}

// ---------------------------------------------------------------------
// Edge MLP:  msg = relu([x[src] | x[dst] | ea] @ ew + eb) -> atomic scatter
// Block = 128 threads = 4 waves; each wave owns a 16-edge tile.
// ---------------------------------------------------------------------
__global__ __launch_bounds__(128) void edge_mlp_wmma(
    const float* __restrict__ x, const int* __restrict__ src,
    const int* __restrict__ dst, const float* __restrict__ ea,
    const _Float16* __restrict__ wpk, const float* __restrict__ eb,
    float* __restrict__ agg, int n_edges)
{
    __shared__ v16h sW[E_KS * E_NT * 32];      // B fragments, fragment-order
    __shared__ v16h sA[4][E_KS * 32];          // per-wave A fragments

    const int tid  = threadIdx.x;
    const int wave = tid >> 5;
    const int lane = tid & 31;

    // bulk-copy pre-swizzled weights: global b128 -> ds b128
    {
        const v16h* gW = (const v16h*)wpk;
        for (int i = tid; i < E_KS * E_NT * 32; i += 128) sW[i] = gW[i];
    }

    // gather A tile as float4 chunks, convert, scatter into fragment order
    const int tile0 = (blockIdx.x * 4 + wave) * 16;
    _Float16* A = (_Float16*)sA[wave];
    for (int i = lane; i < 16 * (E_INP / 4); i += 32) {
        int r = i / (E_INP / 4);
        int k = (i - r * (E_INP / 4)) * 4;
        int e = tile0 + r;
        float4 v = make_float4(0.f, 0.f, 0.f, 0.f);
        if (e < n_edges) {
            if      (k < 64)  v = *(const float4*)(x + src[e] * D_NODE + k);
            else if (k < 128) v = *(const float4*)(x + dst[e] * D_NODE + (k - 64));
            else if (k < 144) v = *(const float4*)(ea + e * D_EDGE + (k - 128));
        }
        v4h hv;
        hv[0] = (_Float16)v.x; hv[1] = (_Float16)v.y;
        hv[2] = (_Float16)v.z; hv[3] = (_Float16)v.w;
        *(v4h*)&A[frag_idx(r, k)] = hv;       // 8B-aligned ds_store_b64
    }
    __syncthreads();

    const int row = lane & 15;
    const int hi  = lane >> 4;
    v8f c0 = {}, c1 = {};

    #pragma unroll
    for (int ks = 0; ks < E_KS; ++ks) {
        v16h af = sA[wave][ks * 32 + lane];
        v16h b0 = sW[(ks * E_NT + 0) * 32 + lane];
        v16h b1 = sW[(ks * E_NT + 1) * 32 + lane];
        c0 = __builtin_amdgcn_wmma_f32_16x16x32_f16(false, af, false, b0,
                                                    (short)0, c0, false, false);
        c1 = __builtin_amdgcn_wmma_f32_16x16x32_f16(false, af, false, b1,
                                                    (short)0, c1, false, false);
    }

    // bias + relu + atomic scatter-add onto agg[src]
    float bias0 = eb[row], bias1 = eb[16 + row];
    #pragma unroll
    for (int r = 0; r < 8; ++r) {
        int m = r + (hi ? 8 : 0);
        int e = tile0 + m;
        if (e < n_edges) {
            int s = src[e];
            float v0 = c0[r] + bias0;
            float v1 = c1[r] + bias1;
            v0 = v0 > 0.f ? v0 : 0.f;
            v1 = v1 > 0.f ? v1 : 0.f;
            unsafeAtomicAdd(&agg[s * MSG + row],      v0);
            unsafeAtomicAdd(&agg[s * MSG + 16 + row], v1);
        }
    }
}

// ---------------------------------------------------------------------
// Node MLP: h = relu([x | agg] @ nw + nb)   [N,96] @ [96,64]
// ---------------------------------------------------------------------
__global__ __launch_bounds__(128) void node_mlp_wmma(
    const float* __restrict__ xin, const float* __restrict__ agg,
    const _Float16* __restrict__ wpk, const float* __restrict__ nb,
    float* __restrict__ xout, int n_nodes)
{
    __shared__ v16h sW[N_KS * N_NT * 32];
    __shared__ v16h sA[4][N_KS * 32];

    const int tid  = threadIdx.x;
    const int wave = tid >> 5;
    const int lane = tid & 31;

    {
        const v16h* gW = (const v16h*)wpk;
        for (int i = tid; i < N_KS * N_NT * 32; i += 128) sW[i] = gW[i];
    }

    const int tile0 = (blockIdx.x * 4 + wave) * 16;
    _Float16* A = (_Float16*)sA[wave];
    for (int i = lane; i < 16 * (N_IN / 4); i += 32) {
        int r = i / (N_IN / 4);
        int k = (i - r * (N_IN / 4)) * 4;
        int n = tile0 + r;
        float4 v = make_float4(0.f, 0.f, 0.f, 0.f);
        if (n < n_nodes) {
            if (k < D_NODE) v = *(const float4*)(xin + n * D_NODE + k);
            else            v = *(const float4*)(agg + n * MSG + (k - D_NODE));
        }
        v4h hv;
        hv[0] = (_Float16)v.x; hv[1] = (_Float16)v.y;
        hv[2] = (_Float16)v.z; hv[3] = (_Float16)v.w;
        *(v4h*)&A[frag_idx(r, k)] = hv;
    }
    __syncthreads();

    const int row = lane & 15;
    const int hi  = lane >> 4;
    v8f c[N_NT] = {{}, {}, {}, {}};

    #pragma unroll
    for (int ks = 0; ks < N_KS; ++ks) {
        v16h af = sA[wave][ks * 32 + lane];
        #pragma unroll
        for (int nt = 0; nt < N_NT; ++nt) {
            v16h bf = sW[(ks * N_NT + nt) * 32 + lane];
            c[nt] = __builtin_amdgcn_wmma_f32_16x16x32_f16(false, af, false, bf,
                                                           (short)0, c[nt], false, false);
        }
    }

    #pragma unroll
    for (int nt = 0; nt < N_NT; ++nt) {
        float bias = nb[nt * 16 + row];
        #pragma unroll
        for (int r = 0; r < 8; ++r) {
            int m = r + (hi ? 8 : 0);
            int n = tile0 + m;
            if (n < n_nodes) {
                float v = c[nt][r] + bias;
                xout[n * D_NODE + nt * 16 + row] = v > 0.f ? v : 0.f;
            }
        }
    }
}

// ---------------------------------------------------------------------
// small helper kernels (head is <0.1% of FLOPs -> scalar fp32)
// ---------------------------------------------------------------------
__global__ void zero_f32(float* p, int n) {
    int i = blockIdx.x * blockDim.x + threadIdx.x;
    if (i < n) p[i] = 0.f;
}

__global__ void pool_sum(const float* __restrict__ h, const int* __restrict__ batch,
                         float* __restrict__ sums, float* __restrict__ cnt, int n_nodes)
{
    int i = blockIdx.x * blockDim.x + threadIdx.x;
    int node = i >> 6, col = i & 63;
    if (node < n_nodes) {
        int b = batch[node];
        unsafeAtomicAdd(&sums[b * D_NODE + col], h[node * D_NODE + col]);
        if (col == 0) unsafeAtomicAdd(&cnt[b], 1.f);
    }
}

__global__ void pool_div(const float* __restrict__ sums, const float* __restrict__ cnt,
                         float* __restrict__ g)
{
    int i = blockIdx.x * blockDim.x + threadIdx.x;
    if (i < N_GRAPHS * D_NODE) g[i] = sums[i] / fmaxf(cnt[i >> 6], 1.f);
}

__global__ void mm_bias(const float* __restrict__ A, const float* __restrict__ W,
                        const float* __restrict__ b, float* __restrict__ C,
                        int M, int K, int N)
{
    int i = blockIdx.x * blockDim.x + threadIdx.x;
    if (i >= M * N) return;
    int m = i / N, n = i - m * N;
    float acc = b[n];
    for (int k = 0; k < K; ++k) acc += A[m * K + k] * W[k * N + n];
    C[i] = acc;
}

__global__ void bn_stats(const float* __restrict__ H, int M, int N,
                         float* __restrict__ mu, float* __restrict__ rs)
{
    __shared__ float ss[256], sq[256];
    int j = blockIdx.x, t = threadIdx.x;
    float s = 0.f, q = 0.f;
    for (int i = t; i < M; i += 256) { float v = H[i * N + j]; s += v; q += v * v; }
    ss[t] = s; sq[t] = q; __syncthreads();
    for (int o = 128; o > 0; o >>= 1) {
        if (t < o) { ss[t] += ss[t + o]; sq[t] += sq[t + o]; }
        __syncthreads();
    }
    if (t == 0) {
        float m = ss[0] / (float)M;
        float v = sq[0] / (float)M - m * m;
        mu[j] = m;
        rs[j] = rsqrtf(v + 1e-5f);
    }
}

__global__ void bn_apply_relu(float* __restrict__ H, const float* __restrict__ mu,
                              const float* __restrict__ rs, const float* __restrict__ g,
                              const float* __restrict__ b, int M, int N)
{
    int i = blockIdx.x * blockDim.x + threadIdx.x;
    if (i >= M * N) return;
    int j = i % N;
    float v = (H[i] - mu[j]) * rs[j] * g[j] + b[j];
    H[i] = v > 0.f ? v : 0.f;
}

// ---------------------------------------------------------------------
extern "C" void kernel_launch(void* const* d_in, const int* in_sizes, int n_in,
                              void* d_out, int out_size, void* d_ws, size_t ws_size,
                              hipStream_t stream)
{
    const float* x   = (const float*)d_in[0];
    const int*   ei  = (const int*)d_in[1];
    const float* ea  = (const float*)d_in[2];
    const int*   bat = (const int*)d_in[3];
    const float *ew1 = (const float*)d_in[4],  *eb1 = (const float*)d_in[5];
    const float *nw1 = (const float*)d_in[6],  *nb1 = (const float*)d_in[7];
    const float *ew2 = (const float*)d_in[8],  *eb2 = (const float*)d_in[9];
    const float *nw2 = (const float*)d_in[10], *nb2 = (const float*)d_in[11];
    const float *ew3 = (const float*)d_in[12], *eb3 = (const float*)d_in[13];
    const float *nw3 = (const float*)d_in[14], *nb3 = (const float*)d_in[15];
    const float *gw1 = (const float*)d_in[16], *gb1 = (const float*)d_in[17];
    const float *bg1 = (const float*)d_in[18], *bb1 = (const float*)d_in[19];
    const float *gw2 = (const float*)d_in[20], *gb2 = (const float*)d_in[21];
    const float *bg2 = (const float*)d_in[22], *bb2 = (const float*)d_in[23];
    const float *gw3 = (const float*)d_in[24], *gb3 = (const float*)d_in[25];

    const int* src = ei;
    const int* dst = ei + N_EDGES;

    // workspace layout (floats)
    float* xA   = (float*)d_ws;                  // [N, 64]
    float* xB   = xA   + N_NODES * D_NODE;       // [N, 64]
    float* agg  = xB   + N_NODES * D_NODE;       // [N, 32]
    float* sums = agg  + N_NODES * MSG;          // [G, 64]
    float* cnt  = sums + N_GRAPHS * D_NODE;      // [G]
    float* h1   = cnt  + N_GRAPHS;               // [G, 128]
    float* h2   = h1   + N_GRAPHS * HID;         // [G, 128]
    float* mu   = h2   + N_GRAPHS * HID;         // [128]
    float* rs   = mu   + HID;                    // [128]
    float* g    = rs   + HID;                    // [G, 64]

    // packed f16 weight region (32B aligned for v16h loads)
    uintptr_t wp = (uintptr_t)(g + N_GRAPHS * D_NODE);
    wp = (wp + 31) & ~(uintptr_t)31;
    _Float16* wbase = (_Float16*)wp;
    const int EWPK = E_KS * E_NT * 32 * 16;      // 5120 halfs (10240B, 32B-mult)
    const int NWPK = N_KS * N_NT * 32 * 16;      // 6144 halfs (12288B, 32B-mult)
    _Float16* wpkE[3] = {wbase, wbase + EWPK, wbase + 2 * EWPK};
    _Float16* wpkN0  = wbase + 3 * EWPK;
    _Float16* wpkN[3] = {wpkN0, wpkN0 + NWPK, wpkN0 + 2 * NWPK};

    const float* EW[3] = {ew1, ew2, ew3}; const float* EB[3] = {eb1, eb2, eb3};
    const float* NW[3] = {nw1, nw2, nw3}; const float* NB[3] = {nb1, nb2, nb3};

    // pre-swizzle weights into fragment order (tiny, once per call)
    for (int l = 0; l < 3; ++l) {
        pack_w<<<(EWPK + 255) / 256, 256, 0, stream>>>(EW[l], wpkE[l], E_IN, MSG, E_NT, EWPK);
        pack_w<<<(NWPK + 255) / 256, 256, 0, stream>>>(NW[l], wpkN[l], N_IN, D_NODE, N_NT, NWPK);
    }

    const int edge_grid = (N_EDGES + 63) / 64;   // 4 tiles of 16 edges per block
    const int node_grid = (N_NODES + 63) / 64;
    const int aggN = N_NODES * MSG;

    const float* xcur = x;
    float* xnxt = xA;
    for (int l = 0; l < 3; ++l) {
        zero_f32<<<(aggN + 255) / 256, 256, 0, stream>>>(agg, aggN);
        edge_mlp_wmma<<<edge_grid, 128, 0, stream>>>(xcur, src, dst, ea,
                                                     wpkE[l], EB[l], agg, N_EDGES);
        node_mlp_wmma<<<node_grid, 128, 0, stream>>>(xcur, agg, wpkN[l], NB[l],
                                                     xnxt, N_NODES);
        xcur = xnxt;
        xnxt = (xnxt == xA) ? xB : xA;
    }

    // mean pooling over graphs
    const int poolN = N_GRAPHS * D_NODE + N_GRAPHS;          // sums + cnt contiguous
    zero_f32<<<(poolN + 255) / 256, 256, 0, stream>>>(sums, poolN);
    pool_sum<<<(N_NODES * D_NODE + 255) / 256, 256, 0, stream>>>(xcur, bat, sums, cnt, N_NODES);
    pool_div<<<(N_GRAPHS * D_NODE + 255) / 256, 256, 0, stream>>>(sums, cnt, g);

    // head MLP with training-mode BatchNorm
    mm_bias<<<(N_GRAPHS * HID + 255) / 256, 256, 0, stream>>>(g, gw1, gb1, h1,
                                                              N_GRAPHS, D_NODE, HID);
    bn_stats<<<HID, 256, 0, stream>>>(h1, N_GRAPHS, HID, mu, rs);
    bn_apply_relu<<<(N_GRAPHS * HID + 255) / 256, 256, 0, stream>>>(h1, mu, rs, bg1, bb1,
                                                                    N_GRAPHS, HID);
    mm_bias<<<(N_GRAPHS * HID + 255) / 256, 256, 0, stream>>>(h1, gw2, gb2, h2,
                                                              N_GRAPHS, HID, HID);
    bn_stats<<<HID, 256, 0, stream>>>(h2, N_GRAPHS, HID, mu, rs);
    bn_apply_relu<<<(N_GRAPHS * HID + 255) / 256, 256, 0, stream>>>(h2, mu, rs, bg2, bb2,
                                                                    N_GRAPHS, HID);
    mm_bias<<<(N_GRAPHS + 255) / 256, 256, 0, stream>>>(h2, gw3, gb3, (float*)d_out,
                                                        N_GRAPHS, HID, 1);
}
